// SplitOffloadedTransformer_18957985644859
// MI455X (gfx1250) — matrix operand order repaired and checked by reading, not compiled, not run on hardware
//
#include <hip/hip_runtime.h>
#include <hip/hip_bf16.h>

// ---- problem constants (match reference) ----
#define LNUM 2
#define BB   2
#define SS   1024
#define DD   2048
#define HH   16
#define HD   128
#define SKV  1024
#define VV   16384
#define BS   (BB*SS)
#define D3   (3*DD)
#define D4   (4*DD)

typedef __bf16 bf16;
typedef __attribute__((ext_vector_type(16))) __bf16 bf16x16;
typedef __attribute__((ext_vector_type(8)))  __bf16 bf16x8;
typedef __attribute__((ext_vector_type(8)))  float  fx8;

__device__ __forceinline__ fx8 wmma_bf16(bf16x16 a, bf16x16 b, fx8 c) {
    return __builtin_amdgcn_wmma_f32_16x16x32_bf16(
        /*neg_a=*/false, a, /*neg_b=*/false, b,
        /*c_mod=*/(short)0, c, /*reuse_a=*/false, /*reuse_b=*/false);
}

__device__ __forceinline__ bf16x16 cvt_f32x16(fx8 lo, fx8 hi) {
    bf16x16 r;
#pragma unroll
    for (int i = 0; i < 8; ++i) { r[i] = (bf16)lo[i]; r[8 + i] = (bf16)hi[i]; }
    return r;
}

__device__ __forceinline__ bf16x16 pack_bf16(bf16x8 lo, bf16x8 hi) {
    bf16x16 r;
#pragma unroll
    for (int i = 0; i < 8; ++i) { r[i] = lo[i]; r[8 + i] = hi[i]; }
    return r;
}

// ---------------- positional encoding: y = x + PE ----------------
__global__ __launch_bounds__(256)
void pos_enc_kernel(const float* __restrict__ x, const int* __restrict__ cur,
                    float* __restrict__ y) {
    int idx = blockIdx.x * 256 + threadIdx.x;          // < BS*DD
    int d = idx & (DD - 1);
    int s = (idx / DD) & (SS - 1);
    int p = d >> 1;
    // freq = exp(-(ln 1e4 / D) * 2p)
    float freq = __expf(-(9.210340371976184f / (float)DD) * (2.0f * (float)p));
    float ang = (float)(cur[0] + s) * freq;
    float pe = (d & 1) ? __cosf(ang) : __sinf(ang);
    y[idx] = x[idx] + pe;
}

// ---------------- layernorm (f32 in -> bf16 out) ----------------
__global__ __launch_bounds__(256)
void layernorm_kernel(const float* __restrict__ X, const float* __restrict__ w,
                      const float* __restrict__ bia, bf16* __restrict__ Y) {
    const int row = blockIdx.x;
    const int t = threadIdx.x;
    const float* x = X + (size_t)row * DD;
    __shared__ float red[256];
    float vals[8];
    float s = 0.f;
#pragma unroll
    for (int i = 0; i < 8; ++i) { vals[i] = x[t + i * 256]; s += vals[i]; }
    red[t] = s; __syncthreads();
    for (int o = 128; o > 0; o >>= 1) { if (t < o) red[t] += red[t + o]; __syncthreads(); }
    float mean = red[0] * (1.0f / DD);
    __syncthreads();
    s = 0.f;
#pragma unroll
    for (int i = 0; i < 8; ++i) { float d = vals[i] - mean; s += d * d; }
    red[t] = s; __syncthreads();
    for (int o = 128; o > 0; o >>= 1) { if (t < o) red[t] += red[t + o]; __syncthreads(); }
    float rs = rsqrtf(red[0] * (1.0f / DD) + 1e-5f);
#pragma unroll
    for (int i = 0; i < 8; ++i) {
        int d = t + i * 256;
        Y[(size_t)row * DD + d] = (bf16)((vals[i] - mean) * rs * w[d] + bia[d]);
    }
}

// ------------- WMMA GEMM: C[M,N] = A[M,K](bf16) * W[N,K]^T (fp32, res/off split) -------------
// Block = 256 threads = 8 waves (4x2). Block tile 128x128, wave tile 32x64.
// A tile (128x32 bf16, 8KB) is double-buffered in LDS and filled by async DMA
// (ASYNCcnt path): tile k+1's DMA is issued before tile k's WMMAs so the copy
// is hidden behind compute. Weights are fp32->bf16 converted in registers
// (they stay L2-resident across the M sweep, so no separate conversion pass).
__global__ __launch_bounds__(256)
void gemm_kernel(const bf16* __restrict__ A, const float* __restrict__ Wres,
                 const float* __restrict__ Woff, float* __restrict__ C,
                 int M, int N, int K, int Nres) {
    __shared__ bf16 ldsA[2][128 * 32];               // ping-pong, 2 x 8 KB
    const int tid = threadIdx.x;
    const int lane = tid & 31;
    const int wave = tid >> 5;                       // 0..7
    const int hl = lane >> 4;
    const int r = lane & 15;
    const int mblk = blockIdx.y * 128;
    const int nblk = blockIdx.x * 128;
    const int wr = wave >> 1;                        // 0..3
    const int wc = wave & 1;                         // 0..1
    const int m0 = mblk + wr * 32;
    const int n0 = nblk + wc * 64;

    // this thread's two async transfers: 16B each, rows 64 apart (16*wave + lane/4, and +64)
    const int linear0 = wave * 64 + lane;            // 0..479, 32 per wave at stride 64
    const int linear1 = linear0 + 32 * 8;            // +256
    const int row0 = linear0 >> 2, ch0 = (linear0 & 3) * 16;
    const int row1 = linear1 >> 2, ch1 = (linear1 & 3) * 16;
    const char* asrc0 = (const char*)(A + (size_t)(mblk + row0) * K) + ch0;
    const char* asrc1 = (const char*)(A + (size_t)(mblk + row1) * K) + ch1;
    const unsigned adst0 = (unsigned)(size_t)&ldsA[0][row0 * 32] + (unsigned)ch0;
    const unsigned adst1 = (unsigned)(size_t)&ldsA[0][row1 * 32] + (unsigned)ch1;
    const unsigned bufstride = (unsigned)(128 * 32 * sizeof(bf16));

    fx8 acc[2][4];
#pragma unroll
    for (int s = 0; s < 2; ++s)
#pragma unroll
        for (int t = 0; t < 4; ++t) acc[s][t] = (fx8){};

    const float* wrow[4];
#pragma unroll
    for (int t = 0; t < 4; ++t) {
        int n = n0 + t * 16 + r;
        wrow[t] = (n < Nres) ? (Wres + (size_t)n * K) : (Woff + (size_t)(n - Nres) * K);
    }

    // prologue: DMA tile 0 into buffer 0
    asm volatile("global_load_async_to_lds_b128 %0, %1, off"
                 :: "v"(adst0), "v"(asrc0 /* k0 = 0 */) : "memory");
    asm volatile("global_load_async_to_lds_b128 %0, %1, off"
                 :: "v"(adst1), "v"(asrc1) : "memory");
    asm volatile("s_wait_asynccnt 0x0" ::: "memory");
    __syncthreads();

    for (int k0 = 0; k0 < K; k0 += 32) {
        const int cur = (k0 >> 5) & 1;
        // ---- issue async DMA for the NEXT 128x32 tile into the other buffer ----
        if (k0 + 32 < K) {
            const char* g0 = asrc0 + (size_t)(k0 + 32) * sizeof(bf16);
            const char* g1 = asrc1 + (size_t)(k0 + 32) * sizeof(bf16);
            unsigned l0 = adst0 + (cur ^ 1) * bufstride;
            unsigned l1 = adst1 + (cur ^ 1) * bufstride;
            asm volatile("global_load_async_to_lds_b128 %0, %1, off"
                         :: "v"(l0), "v"(g0) : "memory");
            asm volatile("global_load_async_to_lds_b128 %0, %1, off"
                         :: "v"(l1), "v"(g1) : "memory");
        }
        // ---- A fragments from current LDS buffer ----
        bf16x16 af[2];
#pragma unroll
        for (int s = 0; s < 2; ++s) {
            const bf16* ar = &ldsA[cur][(wr * 32 + s * 16 + r) * 32];
            af[s] = pack_bf16(*(const bf16x8*)(ar + hl * 8),
                              *(const bf16x8*)(ar + 16 + hl * 8));
        }
        // ---- B fragments (fp32 -> bf16 in regs) + 8 WMMAs ----
#pragma unroll
        for (int t = 0; t < 4; ++t) {
            __builtin_prefetch(wrow[t] + k0 + 256, 0, 1);
            bf16x16 b = cvt_f32x16(*(const fx8*)(wrow[t] + k0 + hl * 8),
                                   *(const fx8*)(wrow[t] + k0 + 16 + hl * 8));
            acc[0][t] = wmma_bf16(af[0], b, acc[0][t]);
            acc[1][t] = wmma_bf16(af[1], b, acc[1][t]);
        }
        // next tile's DMA had the whole compute phase to land
        asm volatile("s_wait_asynccnt 0x0" ::: "memory");
        __syncthreads();
    }
#pragma unroll
    for (int s = 0; s < 2; ++s)
#pragma unroll
        for (int t = 0; t < 4; ++t)
#pragma unroll
            for (int i = 0; i < 8; ++i) {
                int m = m0 + s * 16 + hl * 8 + i;
                C[(size_t)m * N + n0 + t * 16 + r] = acc[s][t][i];
            }
}

// ------------- flash attention over concat(K_res, K_scr) with last-slot override -------------
__global__ __launch_bounds__(32)
void attn_kernel(const float* __restrict__ qkv,    // [B,S,3D], head-major 3*HD per head
                 const float* __restrict__ Kres, const float* __restrict__ Vres,
                 const float* __restrict__ Koff, const float* __restrict__ Voff,
                 float* __restrict__ O,            // [B,S,D]
                 int layer) {
    const int lane = threadIdx.x;
    const int hl = lane >> 4;
    const int r = lane & 15;
    const int q0 = blockIdx.x * 16;
    const int b = blockIdx.y / HH;
    const int h = blockIdx.y % HH;
    const size_t base = (((size_t)layer * BB + b) * HH + h) * (size_t)SKV * HD;
    const float* qnew = qkv + (size_t)(b * SS + SS - 1) * D3 + h * 3 * HD;
    const float SCALE = 0.08838834764831845f;   // 1/sqrt(128)

    __shared__ bf16 ldsV[HD * 32];   // transposed: [hd][key]
    __shared__ bf16 ldsP[16 * 32];   // [q][key]

    // preload Q fragments (16 rows x 128, four K-chunks of 32)
    const float* qp = qkv + (size_t)(b * SS + q0 + r) * D3 + h * 3 * HD;
    bf16x16 qf[4];
#pragma unroll
    for (int c = 0; c < 4; ++c) {
        int k0 = c * 32;
        qf[c] = cvt_f32x16(*(const fx8*)(qp + k0 + hl * 8),
                           *(const fx8*)(qp + k0 + 16 + hl * 8));
    }

    fx8 o[8];
#pragma unroll
    for (int t = 0; t < 8; ++t) o[t] = (fx8){};
    float mi[8], li[8];
#pragma unroll
    for (int i = 0; i < 8; ++i) { mi[i] = -INFINITY; li[i] = 0.f; }

    for (int kt = 0; kt < 2 * SKV; kt += 32) {
        // ---- stage V tile (32 keys x 128) transposed into LDS as bf16 ----
        {
            int gv = kt + lane;
            const float* vp = Vres + base + (size_t)gv * HD;
            const float* vpo = Voff + base + (size_t)(gv - SKV) * HD;
            vp = (gv >= SKV) ? ((gv == 2 * SKV - 1) ? (qnew + 2 * HD) : vpo) : vp;
#pragma unroll
            for (int j = 0; j < HD; j += 8) {
                fx8 vv = *(const fx8*)(vp + j);
#pragma unroll
                for (int e = 0; e < 8; ++e) ldsV[(j + e) * 32 + lane] = (bf16)vv[e];
            }
        }
        // ---- logits: two 16x16 tiles over 32 keys ----
        fx8 s0 = {}, s1 = {};
        {
            int g0 = kt + r, g1 = kt + 16 + r;
            const float* kp0 = Kres + base + (size_t)g0 * HD;
            const float* k0o = Koff + base + (size_t)(g0 - SKV) * HD;
            kp0 = (g0 >= SKV) ? ((g0 == 2 * SKV - 1) ? (qnew + HD) : k0o) : kp0;
            const float* kp1 = Kres + base + (size_t)g1 * HD;
            const float* k1o = Koff + base + (size_t)(g1 - SKV) * HD;
            kp1 = (g1 >= SKV) ? ((g1 == 2 * SKV - 1) ? (qnew + HD) : k1o) : kp1;
#pragma unroll
            for (int c = 0; c < 4; ++c) {
                int k0 = c * 32;
                bf16x16 b0 = cvt_f32x16(*(const fx8*)(kp0 + k0 + hl * 8),
                                        *(const fx8*)(kp0 + k0 + 16 + hl * 8));
                bf16x16 b1 = cvt_f32x16(*(const fx8*)(kp1 + k0 + hl * 8),
                                        *(const fx8*)(kp1 + k0 + 16 + hl * 8));
                s0 = wmma_bf16(qf[c], b0, s0);
                s1 = wmma_bf16(qf[c], b1, s1);
            }
        }
        // ---- online softmax (row = hl*8+i, cols spread over 16 lanes of the half) ----
        float al[8];
#pragma unroll
        for (int i = 0; i < 8; ++i) {
            float a0 = s0[i] * SCALE, a1 = s1[i] * SCALE;
            float rm = fmaxf(a0, a1);
#pragma unroll
            for (int msk = 1; msk < 16; msk <<= 1) rm = fmaxf(rm, __shfl_xor(rm, msk, 32));
            float nm = fmaxf(mi[i], rm);
            al[i] = __expf(mi[i] - nm);
            float p0 = __expf(a0 - nm), p1 = __expf(a1 - nm);
            float ps = p0 + p1;
#pragma unroll
            for (int msk = 1; msk < 16; msk <<= 1) ps += __shfl_xor(ps, msk, 32);
            li[i] = li[i] * al[i] + ps;
            mi[i] = nm;
            ldsP[(hl * 8 + i) * 32 + r] = (bf16)p0;
            ldsP[(hl * 8 + i) * 32 + 16 + r] = (bf16)p1;
        }
        __syncthreads();
#pragma unroll
        for (int t = 0; t < 8; ++t)
#pragma unroll
            for (int i = 0; i < 8; ++i) o[t][i] *= al[i];
        // ---- O += P (16x32) * V (32xHD) ----
        bf16x16 pa = pack_bf16(*(const bf16x8*)&ldsP[r * 32 + hl * 16],
                               *(const bf16x8*)&ldsP[r * 32 + hl * 16 + 8]);
#pragma unroll
        for (int t = 0; t < 8; ++t) {
            bf16x16 vb = *(const bf16x16*)&ldsV[(t * 16 + r) * 32 + hl * 16];
            o[t] = wmma_bf16(pa, vb, o[t]);
        }
        __syncthreads();
    }
    // ---- finalize & store ----
#pragma unroll
    for (int i = 0; i < 8; ++i) {
        float inv = 1.0f / li[i];
        float* orow = O + (size_t)(b * SS + q0 + hl * 8 + i) * DD + h * HD;
#pragma unroll
        for (int t = 0; t < 8; ++t) orow[t * 16 + r] = o[t][i] * inv;
    }
}

// ---------------- elementwise helpers ----------------
__global__ __launch_bounds__(256)
void add_kernel(float* __restrict__ x, const float* __restrict__ d) {
    int idx = blockIdx.x * 256 + threadIdx.x;
    x[idx] += d[idx];
}

__global__ __launch_bounds__(256)
void gelu_kernel(const float* __restrict__ h, bf16* __restrict__ hb) {
    int idx = blockIdx.x * 256 + threadIdx.x;
    float v = h[idx];
    hb[idx] = (bf16)(0.5f * v * (1.0f + erff(v * 0.70710678118654752f)));
}

__global__ __launch_bounds__(256)
void softmax_kernel(float* __restrict__ logits) {
    const int row = blockIdx.x;
    const int t = threadIdx.x;
    float* p = logits + (size_t)row * VV;
    __shared__ float red[256];
    float mx = -INFINITY;
    for (int j = t; j < VV; j += 256) mx = fmaxf(mx, p[j]);
    red[t] = mx; __syncthreads();
    for (int o = 128; o > 0; o >>= 1) { if (t < o) red[t] = fmaxf(red[t], red[t + o]); __syncthreads(); }
    mx = red[0]; __syncthreads();
    float s = 0.f;
    for (int j = t; j < VV; j += 256) s += __expf(p[j] - mx);
    red[t] = s; __syncthreads();
    for (int o = 128; o > 0; o >>= 1) { if (t < o) red[t] += red[t + o]; __syncthreads(); }
    float inv = 1.0f / red[0];
    for (int j = t; j < VV; j += 256) p[j] = __expf(p[j] - mx) * inv;
}

// ---------------- host orchestration ----------------
extern "C" void kernel_launch(void* const* d_in, const int* in_sizes, int n_in,
                              void* d_out, int out_size, void* d_ws, size_t ws_size,
                              hipStream_t stream) {
    (void)in_sizes; (void)n_in; (void)out_size; (void)ws_size;
    const float* x_in     = (const float*)d_in[0];
    const float* ln_w     = (const float*)d_in[1];
    const float* ln_b     = (const float*)d_in[2];
    const float* qkv_res  = (const float*)d_in[3];
    const float* qkv_off  = (const float*)d_in[4];
    const float* k_res    = (const float*)d_in[5];
    const float* v_res    = (const float*)d_in[6];
    const float* k_off    = (const float*)d_in[7];
    const float* v_off    = (const float*)d_in[8];
    const float* ffn1_res = (const float*)d_in[9];
    const float* ffn1_off = (const float*)d_in[10];
    const float* ffn2_res = (const float*)d_in[11];
    const float* ffn2_off = (const float*)d_in[12];
    const float* out_res  = (const float*)d_in[13];
    const float* out_off  = (const float*)d_in[14];
    const int*   cur      = (const int*)d_in[15];
    float* out = (float*)d_out;

    // workspace carve (all chunks 256B aligned by construction)
    char* w = (char*)d_ws;
    float* x    = (float*)w; w += (size_t)BS * DD * 4;   // 16 MB
    bf16*  xn   = (bf16*)w;  w += (size_t)BS * DD * 2;   // 8 MB
    float* qkv  = (float*)w; w += (size_t)BS * D3 * 4;   // 48 MB
    float* tmp  = (float*)w; w += (size_t)BS * DD * 4;   // 16 MB
    float* hbuf = (float*)w; w += (size_t)BS * D4 * 4;   // 64 MB
    bf16*  hb   = (bf16*)w;  w += (size_t)BS * D4 * 2;   // 32 MB

    const int nBSD = BS * DD;
    pos_enc_kernel<<<nBSD / 256, 256, 0, stream>>>(x_in, cur, x);

    for (int l = 0; l < LNUM; ++l) {
        layernorm_kernel<<<BS, 256, 0, stream>>>(x, ln_w, ln_b, xn);
        // QKV: [BS, 3D] = xn * Wqkv^T  (rows 0..3071 res, 3072..6143 off)
        gemm_kernel<<<dim3(D3 / 128, BS / 128), 256, 0, stream>>>(
            xn, qkv_res + (size_t)l * (3 * DD / 2) * DD,
                qkv_off + (size_t)l * (3 * DD / 2) * DD,
            qkv, BS, D3, DD, 3 * DD / 2);
        attn_kernel<<<dim3(SS / 16, BB * HH), 32, 0, stream>>>(
            qkv, k_res, v_res, k_off, v_off, tmp, l);
        add_kernel<<<nBSD / 256, 256, 0, stream>>>(x, tmp);

        layernorm_kernel<<<BS, 256, 0, stream>>>(x, ln_w, ln_b, xn);
        gemm_kernel<<<dim3(D4 / 128, BS / 128), 256, 0, stream>>>(
            xn, ffn1_res + (size_t)l * (2 * DD) * DD,
                ffn1_off + (size_t)l * (2 * DD) * DD,
            hbuf, BS, D4, DD, 2 * DD);
        gelu_kernel<<<(BS * D4) / 256, 256, 0, stream>>>(hbuf, hb);
        gemm_kernel<<<dim3(DD / 128, BS / 128), 256, 0, stream>>>(
            hb, ffn2_res + (size_t)l * (DD / 2) * D4,
                ffn2_off + (size_t)l * (DD / 2) * D4,
            tmp, BS, DD, D4, DD / 2);
        add_kernel<<<nBSD / 256, 256, 0, stream>>>(x, tmp);
    }

    layernorm_kernel<<<BS, 256, 0, stream>>>(x, ln_w, ln_b, xn);
    gemm_kernel<<<dim3(VV / 128, BS / 128), 256, 0, stream>>>(
        xn, out_res, out_off, out, BS, VV, DD, VV / 2);
    softmax_kernel<<<BS, 256, 0, stream>>>(out);
}
